// ContextualAttention_Enhance_30382598652158
// MI455X (gfx1250) — matrix operand order repaired and checked by reading, not compiled, not run on hardware
//
#include <hip/hip_runtime.h>

// ---------------- problem constants ----------------
#define BATCH 2
#define CCH   64
#define HH    96
#define WW    96
#define KSZ   7
#define L1    576      // 24*24 stride-4 patches
#define L2    9216     // 96*96 stride-1 patches
#define DD    3136     // 64*7*7
#define OH1   24
#define P1H   99       // stride-4 padding: top/left 1, bottom/right 2
#define P1W   99
#define P2H   102      // stride-1 padding: 3 each side
#define P2W   102
#define LDSS  40       // LDS row stride in bf16 elems (80B: 16B-aligned, conflict-free)

typedef __attribute__((ext_vector_type(16))) __bf16 v16bf;
typedef __attribute__((ext_vector_type(8)))  float  v8f;

union Frag16 {
  v16bf v;
  unsigned short h[16];
};
union Pack8 {
  unsigned short h[8];
  uint4 v;
};

__device__ __forceinline__ unsigned short f2bf(float f) {
  unsigned u = __float_as_uint(f);
  unsigned r = u + 0x7fffu + ((u >> 16) & 1u);   // round-to-nearest-even
  return (unsigned short)(r >> 16);
}
__device__ __forceinline__ float bf2f(unsigned short h) {
  return __uint_as_float(((unsigned)h) << 16);
}

// ---------------- 0: im2col offset table: d -> (c*102 + i)*102 + j ----------
__global__ void k_offd(int* __restrict__ offD) {
  int d = blockIdx.x * 256 + threadIdx.x;
  if (d < DD) {
    int c = d / 49;
    int r = d - c * 49;
    int i = r / 7;
    int j = r - i * 7;
    offD[d] = c * (P2H * P2W) + i * P2W + j;
  }
}

// ---------------- 1: three 1x1 convs + leaky relu, write padded bf16 maps ----
__global__ void k_conv1x1_lrelu(const float* __restrict__ b_in,
                                const float* __restrict__ g_w, const float* __restrict__ g_b,
                                const float* __restrict__ t_w, const float* __restrict__ t_b,
                                const float* __restrict__ p_w, const float* __restrict__ p_b,
                                unsigned short* __restrict__ b1p,
                                unsigned short* __restrict__ b2p,
                                unsigned short* __restrict__ b3p) {
  int idx = blockIdx.x;
  int b   = idx / (HH * WW);
  int pix = idx - b * HH * WW;
  int y = pix / WW;
  int x = pix - y * WW;
  __shared__ float sh[CCH];
  int t = threadIdx.x;
  sh[t] = b_in[((size_t)(b * CCH + t) * HH + y) * WW + x];
  __syncthreads();
  float a1 = g_b[t], a2 = t_b[t], a3 = p_b[t];
#pragma unroll 8
  for (int c = 0; c < CCH; ++c) {
    float s = sh[c];
    a1 += g_w[t * CCH + c] * s;
    a2 += t_w[t * CCH + c] * s;
    a3 += p_w[t * CCH + c] * s;
  }
  a1 = a1 >= 0.f ? a1 : 0.1f * a1;
  a2 = a2 >= 0.f ? a2 : 0.1f * a2;
  a3 = a3 >= 0.f ? a3 : 0.1f * a3;
  b1p[((size_t)(b * CCH + t) * P1H + (y + 1)) * P1W + (x + 1)] = f2bf(a1);
  b2p[((size_t)(b * CCH + t) * P2H + (y + 3)) * P2W + (x + 3)] = f2bf(a2);
  b3p[((size_t)(b * CCH + t) * P2H + (y + 3)) * P2W + (x + 3)] = f2bf(a3);
}

// ---------------- 2: per (batch, j) sum of squares over (p, c, i) -----------
__global__ void k_nrm(const unsigned short* __restrict__ b1p,
                      float* __restrict__ nrm2) {
  int blk = blockIdx.x;          // b*7 + j
  int b = blk / KSZ;
  int j = blk - b * KSZ;
  __shared__ float red[256];
  float s = 0.f;
  const int N = L1 * CCH * KSZ;
  for (int idx = threadIdx.x; idx < N; idx += 256) {
    int p   = idx / (CCH * KSZ);
    int rem = idx - p * (CCH * KSZ);
    int c = rem / KSZ;
    int i = rem - c * KSZ;
    int py = p / OH1;
    int px = p - py * OH1;
    float v = bf2f(b1p[((size_t)(b * CCH + c) * P1H + py * 4 + i) * P1W + px * 4 + j]);
    s += v * v;
  }
  red[threadIdx.x] = s;
  __syncthreads();
  for (int st = 128; st > 0; st >>= 1) {
    if (threadIdx.x < st) red[threadIdx.x] += red[threadIdx.x + st];
    __syncthreads();
  }
  if (threadIdx.x == 0) nrm2[blk] = red[0];
}

// ---------------- 3: build normalized key matrix w_n (bf16, row-major) ------
__global__ void k_wn(const unsigned short* __restrict__ b1p,
                     const float* __restrict__ nrm2,
                     unsigned short* __restrict__ wn) {
  int blk = blockIdx.x;          // b*L1 + p
  int b = blk / L1;
  int p = blk - b * L1;
  int py = p / OH1, px = p - py * OH1;
  __shared__ float scale[KSZ];
  if (threadIdx.x < KSZ) {
    float n = sqrtf(nrm2[b * KSZ + threadIdx.x]);
    scale[threadIdx.x] = 1.0f / fmaxf(n, 1e-4f);
  }
  __syncthreads();
  for (int d = threadIdx.x; d < DD; d += 256) {
    int c = d / 49;
    int r = d - c * 49;
    int i = r / 7;
    int j = r - i * 7;
    float v = bf2f(b1p[((size_t)(b * CCH + c) * P1H + py * 4 + i) * P1W + px * 4 + j]);
    wn[(size_t)blk * DD + d] = f2bf(v * scale[j]);
  }
}

// ---------------- 4: WMMA GEMM1 scores = w_n . xf^T (double-buffered) -------
__device__ __forceinline__ void g1_gather(const unsigned short* __restrict__ qbase,
                                          const int* __restrict__ op,
                                          Pack8& p0, Pack8& p1) {
#pragma unroll
  for (int e = 0; e < 8; ++e) p0.h[e] = qbase[op[e]];
#pragma unroll
  for (int e = 0; e < 8; ++e) p1.h[e] = qbase[op[8 + e]];
}

__global__ void __launch_bounds__(256) k_gemm_scores(
    const unsigned short* __restrict__ wn,
    const unsigned short* __restrict__ b3p,
    const int* __restrict__ offD,
    float* __restrict__ scores) {
  __shared__ unsigned short At[2][64 * LDSS];    // 2 x  5120 B
  __shared__ unsigned short Bt[2][128 * LDSS];   // 2 x 10240 B, [n_local][k_local]
  int batch = blockIdx.z;
  int n0 = blockIdx.x * 128;
  int m0 = blockIdx.y * 64;
  int tid  = threadIdx.x;
  int wv   = tid >> 5, lane = tid & 31;
  int mw   = wv & 3,   nw   = wv >> 2;
  int half = lane >> 4, lan = lane & 15;

  // staging roles
  int sm = tid >> 2, sc = tid & 3;            // A: row sm, 8-elem chunk sc
  int bn = tid & 127, bkh = tid >> 7;         // B: col bn, 16-elem k-half bkh
  const unsigned short* arow_g = wn + (size_t)(batch * L1 + m0 + sm) * DD;
  const unsigned short* xb = b3p + (size_t)batch * CCH * P2H * P2W;
  int q = n0 + bn;
  int qy = q / WW, qx = q - qy * WW;
  const unsigned short* qbase = xb + qy * P2W + qx;

  const int aoff_st = sm * LDSS + sc * 8;         // staging store offsets
  const int boff_st = bn * LDSS + bkh * 16;
  const int aoff_ld = (mw * 16 + lan) * LDSS + half * 16;    // fragment read offsets
  const int boff_ld = (nw * 64 + lan) * LDSS + half * 16;

  // prologue: stage tile 0 into buffer 0
  {
    uint4 aS = *(const uint4*)(arow_g + aoff_st - sm * LDSS + 0 + sc * 8);
    // (that reads arow_g + sc*16; overwritten below by arow_g + sc*8)
    aS = *(const uint4*)(arow_g + sc * 8);
    Pack8 s0, s1;
    g1_gather(qbase, offD + bkh * 16, s0, s1);
    *(uint4*)(&At[0][aoff_st])     = aS;
    *(uint4*)(&Bt[0][boff_st])     = s0.v;
    *(uint4*)(&Bt[0][boff_st + 8]) = s1.v;
  }
  __syncthreads();

  v8f acc[4] = {};
  for (int kk = 0; kk < DD; kk += 32) {
    int cur = (kk >> 5) & 1;
    bool hasNext = (kk + 32) < DD;
    // (a) issue next tile's global loads early (latency hides under WMMA)
    uint4 aN = {};
    Pack8 nb0, nb1;
    if (hasNext) {
      aN = *(const uint4*)(arow_g + kk + 32 + sc * 8);
      g1_gather(qbase, offD + kk + 32 + bkh * 16, nb0, nb1);
      if (kk + 64 < DD) __builtin_prefetch(arow_g + kk + 64 + sc * 8, 0, 1);
    }
    // (b) load all fragments, then 4 WMMAs
    const unsigned short* ab = At[cur];
    const unsigned short* bb = Bt[cur];
    Frag16 a, b0, b1, b2, b3;
    *(uint4*)&a.h[0]  = *(const uint4*)(ab + aoff_ld);
    *(uint4*)&a.h[8]  = *(const uint4*)(ab + aoff_ld + 8);
    *(uint4*)&b0.h[0] = *(const uint4*)(bb + boff_ld);
    *(uint4*)&b0.h[8] = *(const uint4*)(bb + boff_ld + 8);
    *(uint4*)&b1.h[0] = *(const uint4*)(bb + boff_ld + 16 * LDSS);
    *(uint4*)&b1.h[8] = *(const uint4*)(bb + boff_ld + 16 * LDSS + 8);
    *(uint4*)&b2.h[0] = *(const uint4*)(bb + boff_ld + 32 * LDSS);
    *(uint4*)&b2.h[8] = *(const uint4*)(bb + boff_ld + 32 * LDSS + 8);
    *(uint4*)&b3.h[0] = *(const uint4*)(bb + boff_ld + 48 * LDSS);
    *(uint4*)&b3.h[8] = *(const uint4*)(bb + boff_ld + 48 * LDSS + 8);
    acc[0] = __builtin_amdgcn_wmma_f32_16x16x32_bf16(false, a.v, false, b0.v, (short)0, acc[0], false, false);
    acc[1] = __builtin_amdgcn_wmma_f32_16x16x32_bf16(false, a.v, false, b1.v, (short)0, acc[1], false, false);
    acc[2] = __builtin_amdgcn_wmma_f32_16x16x32_bf16(false, a.v, false, b2.v, (short)0, acc[2], false, false);
    acc[3] = __builtin_amdgcn_wmma_f32_16x16x32_bf16(false, a.v, false, b3.v, (short)0, acc[3], false, false);
    // (c) write staged registers into the other buffer
    if (hasNext) {
      int nxt = cur ^ 1;
      *(uint4*)(&At[nxt][aoff_st])     = aN;
      *(uint4*)(&Bt[nxt][boff_st])     = nb0.v;
      *(uint4*)(&Bt[nxt][boff_st + 8]) = nb1.v;
    }
    // (d) one barrier per iteration covers both buffer hazards
    __syncthreads();
  }
  float* sb = scores + (size_t)batch * L1 * L2;
#pragma unroll
  for (int s = 0; s < 4; ++s) {
    int n = n0 + nw * 64 + s * 16 + lan;
#pragma unroll
    for (int r = 0; r < 8; ++r) {
      int m = m0 + mw * 16 + half * 8 + r;                 // f32 C/D layout
      sb[(size_t)m * L2 + n] = acc[s][r];
    }
  }
}

// ---------------- 5: row softmax (scale 10), emit bf16 attention ------------
__global__ void k_softmax(const float* __restrict__ scores,
                          unsigned short* __restrict__ att) {
  int row = blockIdx.x;          // b*L1 + p
  const float* s = scores + (size_t)row * L2;
  __shared__ float red[256];
  float mx = -1e30f;
  for (int q = threadIdx.x; q < L2; q += 256) mx = fmaxf(mx, s[q]);
  red[threadIdx.x] = mx;
  __syncthreads();
  for (int st = 128; st > 0; st >>= 1) {
    if (threadIdx.x < st) red[threadIdx.x] = fmaxf(red[threadIdx.x], red[threadIdx.x + st]);
    __syncthreads();
  }
  mx = red[0];
  __syncthreads();
  float sum = 0.f;
  for (int q = threadIdx.x; q < L2; q += 256) sum += __expf((s[q] - mx) * 10.0f);
  red[threadIdx.x] = sum;
  __syncthreads();
  for (int st = 128; st > 0; st >>= 1) {
    if (threadIdx.x < st) red[threadIdx.x] += red[threadIdx.x + st];
    __syncthreads();
  }
  float inv = 1.0f / red[0];
  for (int q = threadIdx.x; q < L2; q += 256)
    att[(size_t)row * L2 + q] = f2bf(__expf((s[q] - mx) * 10.0f) * inv);
}

// ---------------- 6: WMMA GEMM2 y_p = att . pf (double-buffered) ------------
__device__ __forceinline__ void g2_gather(const unsigned short* __restrict__ pbase,
                                          int q0, Pack8& p0, Pack8& p1) {
  int qy = q0 / WW, qx = q0 - (q0 / WW) * WW;
#pragma unroll
  for (int e = 0; e < 8; ++e) {
    p0.h[e] = pbase[qy * P2W + qx];
    if (++qx == WW) { qx = 0; ++qy; }
  }
#pragma unroll
  for (int e = 0; e < 8; ++e) {
    p1.h[e] = pbase[qy * P2W + qx];
    if (++qx == WW) { qx = 0; ++qy; }
  }
}

__global__ void __launch_bounds__(128) k_gemm_yp(
    const unsigned short* __restrict__ att,
    const unsigned short* __restrict__ b2p,
    const int* __restrict__ offD,
    float* __restrict__ yp) {
  __shared__ unsigned short At[2][64 * LDSS];    // 2 x 5120 B
  __shared__ unsigned short Bt[2][64 * LDSS];    // 2 x 5120 B, [n_local=d][k_local=q]
  int batch = blockIdx.z;
  int n0 = blockIdx.x * 64;
  int m0 = blockIdx.y * 64;
  int tid  = threadIdx.x;
  int mw   = tid >> 5;
  int lane = tid & 31;
  int half = lane >> 4, lan = lane & 15;

  // staging roles (128 threads)
  int sm = tid >> 1, sc2 = tid & 1;           // A: row sm, 16-elem chunk sc2
  int bn = tid >> 1, bkh = tid & 1;           // B: col bn (=d_local), k-half bkh
  const unsigned short* arow_g = att + (size_t)(batch * L1 + m0 + sm) * L2;
  const unsigned short* vb = b2p + (size_t)batch * CCH * P2H * P2W;
  const unsigned short* pbase = vb + offD[n0 + bn];   // fixed (c,i,j) per thread

  const int aoff_st = sm * LDSS + sc2 * 16;
  const int boff_st = bn * LDSS + bkh * 16;
  const int aoff_ld = (mw * 16 + lan) * LDSS + half * 16;
  const int boff_ld = lan * LDSS + half * 16;

  // prologue: stage tile 0 into buffer 0
  {
    uint4 a0 = *(const uint4*)(arow_g + sc2 * 16);
    uint4 a1 = *(const uint4*)(arow_g + sc2 * 16 + 8);
    Pack8 s0, s1;
    g2_gather(pbase, bkh * 16, s0, s1);
    *(uint4*)(&At[0][aoff_st])     = a0;
    *(uint4*)(&At[0][aoff_st + 8]) = a1;
    *(uint4*)(&Bt[0][boff_st])     = s0.v;
    *(uint4*)(&Bt[0][boff_st + 8]) = s1.v;
  }
  __syncthreads();

  v8f acc[4] = {};
  for (int kk = 0; kk < L2; kk += 32) {
    int cur = (kk >> 5) & 1;
    bool hasNext = (kk + 32) < L2;
    uint4 aN0 = {}, aN1 = {};
    Pack8 nb0, nb1;
    if (hasNext) {
      aN0 = *(const uint4*)(arow_g + kk + 32 + sc2 * 16);
      aN1 = *(const uint4*)(arow_g + kk + 32 + sc2 * 16 + 8);
      g2_gather(pbase, kk + 32 + bkh * 16, nb0, nb1);
      if (kk + 64 < L2) __builtin_prefetch(arow_g + kk + 64 + sc2 * 16, 0, 1);
    }
    const unsigned short* ab = At[cur];
    const unsigned short* bb = Bt[cur];
    Frag16 a, b0, b1, b2, b3;
    *(uint4*)&a.h[0]  = *(const uint4*)(ab + aoff_ld);
    *(uint4*)&a.h[8]  = *(const uint4*)(ab + aoff_ld + 8);
    *(uint4*)&b0.h[0] = *(const uint4*)(bb + boff_ld);
    *(uint4*)&b0.h[8] = *(const uint4*)(bb + boff_ld + 8);
    *(uint4*)&b1.h[0] = *(const uint4*)(bb + boff_ld + 16 * LDSS);
    *(uint4*)&b1.h[8] = *(const uint4*)(bb + boff_ld + 16 * LDSS + 8);
    *(uint4*)&b2.h[0] = *(const uint4*)(bb + boff_ld + 32 * LDSS);
    *(uint4*)&b2.h[8] = *(const uint4*)(bb + boff_ld + 32 * LDSS + 8);
    *(uint4*)&b3.h[0] = *(const uint4*)(bb + boff_ld + 48 * LDSS);
    *(uint4*)&b3.h[8] = *(const uint4*)(bb + boff_ld + 48 * LDSS + 8);
    acc[0] = __builtin_amdgcn_wmma_f32_16x16x32_bf16(false, a.v, false, b0.v, (short)0, acc[0], false, false);
    acc[1] = __builtin_amdgcn_wmma_f32_16x16x32_bf16(false, a.v, false, b1.v, (short)0, acc[1], false, false);
    acc[2] = __builtin_amdgcn_wmma_f32_16x16x32_bf16(false, a.v, false, b2.v, (short)0, acc[2], false, false);
    acc[3] = __builtin_amdgcn_wmma_f32_16x16x32_bf16(false, a.v, false, b3.v, (short)0, acc[3], false, false);
    if (hasNext) {
      int nxt = cur ^ 1;
      *(uint4*)(&At[nxt][aoff_st])     = aN0;
      *(uint4*)(&At[nxt][aoff_st + 8]) = aN1;
      *(uint4*)(&Bt[nxt][boff_st])     = nb0.v;
      *(uint4*)(&Bt[nxt][boff_st + 8]) = nb1.v;
    }
    __syncthreads();
  }
  float* yb = yp + (size_t)batch * L1 * DD;
#pragma unroll
  for (int s = 0; s < 4; ++s) {
    int n = n0 + s * 16 + lan;
#pragma unroll
    for (int r = 0; r < 8; ++r) {
      int m = m0 + mw * 16 + half * 8 + r;
      yb[(size_t)m * DD + n] = acc[s][r];
    }
  }
}

// ---------------- 7: fold (overlap-add / count) + 1x1 conv + residual -------
__global__ void k_fold_conv(const float* __restrict__ yp,
                            const float* __restrict__ b_in,
                            const float* __restrict__ w_w,
                            const float* __restrict__ w_b,
                            float* __restrict__ out) {
  int idx = blockIdx.x;
  int b   = idx / (HH * WW);
  int pix = idx - b * HH * WW;
  int y = pix / WW;
  int x = pix - y * WW;
  int c = threadIdx.x;
  __shared__ float sh[CCH];
  int ypp = y + 3, xpp = x + 3;                 // fold pad = (K-1)/2 = 3
  int py_lo = max(0, (ypp - 3) >> 2);
  int py_hi = min(OH1 - 1, ypp >> 2);
  int px_lo = max(0, (xpp - 3) >> 2);
  int px_hi = min(OH1 - 1, xpp >> 2);
  float acc = 0.f;
  for (int py = py_lo; py <= py_hi; ++py) {
    int i = ypp - 4 * py;
    for (int px = px_lo; px <= px_hi; ++px) {
      int j = xpp - 4 * px;
      int p = py * OH1 + px;
      acc += yp[((size_t)(b * L1 + p)) * DD + c * 49 + i * 7 + j];
    }
  }
  int cnt = (py_hi - py_lo + 1) * (px_hi - px_lo + 1);
  sh[c] = acc / (float)cnt;
  __syncthreads();
  float r = w_b[c];
#pragma unroll 8
  for (int cc = 0; cc < CCH; ++cc) r += w_w[c * CCH + cc] * sh[cc];
  size_t oi = ((size_t)(b * CCH + c) * HH + y) * WW + x;
  out[oi] = b_in[oi] + r;
}

// ---------------- host: workspace carve + launches --------------------------
extern "C" void kernel_launch(void* const* d_in, const int* in_sizes, int n_in,
                              void* d_out, int out_size, void* d_ws, size_t ws_size,
                              hipStream_t stream) {
  (void)in_sizes; (void)n_in; (void)out_size; (void)ws_size;
  const float* b_in = (const float*)d_in[0];
  const float* g_w  = (const float*)d_in[1];
  const float* g_b  = (const float*)d_in[2];
  const float* t_w  = (const float*)d_in[3];
  const float* t_b  = (const float*)d_in[4];
  const float* p_w  = (const float*)d_in[5];
  const float* p_b  = (const float*)d_in[6];
  const float* w_w  = (const float*)d_in[7];
  const float* w_b  = (const float*)d_in[8];
  float* out = (float*)d_out;

  char* base = (char*)d_ws;
  size_t off = 0;
  auto carve = [&](size_t bytes) -> char* {
    char* p = base + off;
    off += (bytes + 255) & ~(size_t)255;
    return p;
  };
  const size_t b1p_bytes = (size_t)BATCH * CCH * P1H * P1W * 2;
  const size_t b2p_bytes = (size_t)BATCH * CCH * P2H * P2W * 2;
  unsigned short* b1p    = (unsigned short*)carve(b1p_bytes);
  unsigned short* b2p    = (unsigned short*)carve(b2p_bytes);
  unsigned short* b3p    = (unsigned short*)carve(b2p_bytes);
  float*          nrm2   = (float*)carve((size_t)BATCH * KSZ * sizeof(float));
  int*            offD   = (int*)carve((size_t)DD * sizeof(int));
  unsigned short* wn     = (unsigned short*)carve((size_t)BATCH * L1 * DD * 2);
  float*          scores = (float*)carve((size_t)BATCH * L1 * L2 * 4);
  unsigned short* att    = (unsigned short*)carve((size_t)BATCH * L1 * L2 * 2);
  float*          yp     = (float*)carve((size_t)BATCH * L1 * DD * 4);

  hipMemsetAsync(b1p, 0, b1p_bytes, stream);
  hipMemsetAsync(b2p, 0, b2p_bytes, stream);
  hipMemsetAsync(b3p, 0, b2p_bytes, stream);

  k_offd<<<(DD + 255) / 256, 256, 0, stream>>>(offD);
  k_conv1x1_lrelu<<<BATCH * HH * WW, CCH, 0, stream>>>(
      b_in, g_w, g_b, t_w, t_b, p_w, p_b, b1p, b2p, b3p);
  k_nrm<<<BATCH * KSZ, 256, 0, stream>>>(b1p, nrm2);
  k_wn<<<BATCH * L1, 256, 0, stream>>>(b1p, nrm2, wn);
  k_gemm_scores<<<dim3(L2 / 128, L1 / 64, BATCH), 256, 0, stream>>>(wn, b3p, offD, scores);
  k_softmax<<<BATCH * L1, 256, 0, stream>>>(scores, att);
  k_gemm_yp<<<dim3(DD / 64, L1 / 64, BATCH), 128, 0, stream>>>(att, b2p, offD, yp);
  k_fold_conv<<<BATCH * HH * WW, CCH, 0, stream>>>(yp, b_in, w_w, w_b, out);
}
